// DGM_77318001263213
// MI455X (gfx1250) — matrix operand order repaired and checked by reading, not compiled, not run on hardware
//
#include <hip/hip_runtime.h>
#include <stdint.h>

typedef unsigned short u16;
typedef __attribute__((ext_vector_type(16))) __bf16 v16bf;
typedef __attribute__((ext_vector_type(8)))  float  v8f;

struct alignas(16) U4 { uint32_t x, y, z, w; };
struct alignas(16) FragBits { U4 lo, hi; };

__device__ __forceinline__ u16 f2bf(float f) {
  uint32_t u = __float_as_uint(f);
  uint32_t r = u + 0x7FFFu + ((u >> 16) & 1u);   // round-to-nearest-even
  return (u16)(r >> 16);
}
__device__ __forceinline__ float bf2f(u16 b) {
  return __uint_as_float(((uint32_t)b) << 16);
}
__device__ __forceinline__ v16bf make_frag(U4 lo, U4 hi) {
  FragBits f{lo, hi};
  return __builtin_bit_cast(v16bf, f);
}

// ---- async global -> LDS (CDNA5 ASYNCcnt path) ----
__device__ __forceinline__ void async_chunk_b128(uint32_t ldsoff, const u16* g) {
  asm volatile("global_load_async_to_lds_b128 %0, %1, off" :: "v"(ldsoff), "v"(g) : "memory");
}
__device__ __forceinline__ void wait_async_le8() {
  asm volatile("s_wait_asynccnt 0x8" ::: "memory");
}
__device__ __forceinline__ void wait_async_0() {
  asm volatile("s_wait_asynccnt 0x0" ::: "memory");
}
__device__ __forceinline__ uint32_t lds_off(const void* p) {
  return (uint32_t)(uintptr_t)p;   // LDS generic addr: offset in addr[31:0]
}

// A-operand (16x32, M x K) direct from global row-major [M][ld]
__device__ __forceinline__ v16bf load_a(const u16* p, int ld) {
  int lane = threadIdx.x & 31;
  int r  = lane & 15;
  int ko = (lane >> 4) << 4;
  const u16* q = p + (size_t)r * ld + ko;
  return make_frag(*(const U4*)q, *(const U4*)(q + 8));
}
// B-operand (32x16, K x N) from B^T storage [N][ld]
__device__ __forceinline__ v16bf load_b(const u16* p, int ld) {
  int lane = threadIdx.x & 31;
  int n  = lane & 15;
  int kh = (lane >> 4) << 4;
  const u16* q = p + (size_t)n * ld + kh;
  return make_frag(*(const U4*)q, *(const U4*)(q + 8));
}
// Fragment loads from an LDS tile [128][72] (K-tile 64, padded stride vs bank conflicts)
__device__ __forceinline__ v16bf lds_frag_a64(const u16 (*s)[72], int r0, int kc) {
  int lane = threadIdx.x & 31;
  const u16* q = &s[r0 + (lane & 15)][kc + ((lane >> 4) << 4)];
  return make_frag(*(const U4*)q, *(const U4*)(q + 8));
}
__device__ __forceinline__ v16bf lds_frag_b64(const u16 (*s)[72], int r0, int kc) {
  int lane = threadIdx.x & 31;
  const u16* q = &s[r0 + (lane & 15)][kc + ((lane >> 4) << 4)];
  return make_frag(*(const U4*)q, *(const U4*)(q + 8));
}
__device__ __forceinline__ v8f wmma_bf16(v16bf a, v16bf b, v8f c) {
  return __builtin_amdgcn_wmma_f32_16x16x32_bf16(false, a, false, b, (short)0, c, false, false);
}

// 128x64 u16 tile from row-major source -> LDS via async DMA (4 b128/thread, 8/wave)
__device__ __forceinline__ void tile_rows_async64(u16 (*dst)[72], const u16* src, int ld_src,
                                                  int kb, int tid) {
  for (int i = 0; i < 4; ++i) {
    int id = tid + i * 256;
    int row = id >> 3, ko = (id & 7) << 3;
    async_chunk_b128(lds_off(&dst[row][ko]), src + (size_t)row * ld_src + kb + ko);
  }
}
// 128x64 u16 tile of implicit-GEMM conv patches (2x2 s2) -> LDS via async DMA
__device__ __forceinline__ void tile_patch_async64(u16 (*dst)[72], const u16* xbb, int n0,
                                                   int kb, int tid) {
  int p = kb >> 8, ky = p >> 1, kx = p & 1;    // kb multiple of 64: one (ky,kx) per stage
  for (int i = 0; i < 4; ++i) {
    int id = tid + i * 256;
    int col = id >> 3, ko = (id & 7) << 3;
    int s = n0 + col;
    int oh = s >> 5, ow = s & 31;
    const u16* g = xbb + (size_t)((((oh * 2 + ky) << 6) + (ow * 2 + kx))) * 256 + (kb & 255) + ko;
    async_chunk_b128(lds_off(&dst[col][ko]), g);
  }
}

// ---------------- Stage 0: x (NCHW f32) -> xb (BHWC bf16) ----------------
__global__ void k_x_to_hwc(const float* __restrict__ x, u16* __restrict__ xb) {
  int idx = blockIdx.x * blockDim.x + threadIdx.x;     // over 16*4096*256
  int c = idx & 255;
  int s = (idx >> 8) & 4095;
  int b = idx >> 20;
  xb[idx] = f2bf(x[(((size_t)b << 8) + c) * 4096 + s]);
}

// dc_w [12,256,3,3] -> dcw [16][2304], K=(ky*3+kx)*256+c, rows 12..15 zero
__global__ void k_prep_dcw(const float* __restrict__ dc_w, u16* __restrict__ dcw) {
  int idx = blockIdx.x * blockDim.x + threadIdx.x;     // 16*2304
  int kk = idx % 2304, m = idx / 2304;
  int p = kk >> 8, c = kk & 255;
  int ky = p / 3, kx = p % 3;
  float v = (m < 12) ? dc_w[(((size_t)m * 256 + c) * 3 + ky) * 3 + kx] : 0.f;
  dcw[idx] = f2bf(v);
}
// v_w [1024,256,2,2] -> vw [1024][1024], K=(ky*2+kx)*256+c
__global__ void k_prep_vw(const float* __restrict__ v_w, u16* __restrict__ vw) {
  int idx = blockIdx.x * blockDim.x + threadIdx.x;     // 1024*1024
  int kk = idx & 1023, oc = idx >> 10;
  int p = kk >> 8, c = kk & 255;
  int ky = p >> 1, kx = p & 1;
  vw[idx] = f2bf(v_w[(((size_t)oc * 256 + c) * 2 + ky) * 2 + kx]);
}

// ---------------- Stage 1: dir_map = conv3x3 SAME (implicit GEMM, M=16) ----------------
__global__ void k_dc_conv(const u16* __restrict__ xb, const u16* __restrict__ dcw,
                          const float* __restrict__ dc_b, u16* __restrict__ dmap) {
  int b = blockIdx.z;
  int wave = threadIdx.x >> 5, lane = threadIdx.x & 31;
  int col = blockIdx.x * 128 + wave * 16 + (lane & 15);
  int h = col >> 6, w = col & 63;
  int kh = (lane >> 4) << 4;
  const u16* xbb = xb + (((size_t)b) << 12) * 256;
  v8f acc = {};
  for (int kb = 0; kb < 2304; kb += 32) {
    v16bf af = load_a(dcw + kb, 2304);
    int p = kb >> 8;
    int ky = p / 3, kx = p % 3;
    int ih = h + ky - 1, iw = w + kx - 1;
    U4 lo = {}, hi = {};
    if (ih >= 0 && ih < 64 && iw >= 0 && iw < 64) {
      const u16* q = xbb + (size_t)((ih << 6) + iw) * 256 + (kb & 255) + kh;
      lo = *(const U4*)q; hi = *(const U4*)(q + 8);
    }
    acc = wmma_bf16(af, make_frag(lo, hi), acc);
  }
  int mbase = (lane >> 4) << 3;
  u16* dst = dmap + ((size_t)((b << 12) + col) << 4);
  for (int g = 0; g < 8; ++g) {
    int m = mbase + g;
    float bias = (m < 12) ? dc_b[m] : 0.f;
    dst[m] = f2bf(acc[g] + bias);
  }
}

// ---------------- Stage 2: q/k conv 2x2 s2 (tiny) -> qt/kt [b][1024][64] bf16 ----------------
__global__ void k_qk_conv(const u16* __restrict__ dmap,
                          const float* __restrict__ q_w, const float* __restrict__ q_b,
                          const float* __restrict__ k_w, const float* __restrict__ k_b,
                          u16* __restrict__ qt, u16* __restrict__ kt) {
  int idx = blockIdx.x * blockDim.x + threadIdx.x;   // 16*1024*2*64
  int oc = idx & 63;
  int which = (idx >> 6) & 1;
  int s = (idx >> 7) & 1023;
  int b = idx >> 17;
  const float* wgt  = which ? k_w : q_w;
  const float* bias = which ? k_b : q_b;
  u16* out = which ? kt : qt;
  float acc = 0.f;
  if (oc < 48) {
    int oh = s >> 5, ow = s & 31;
    acc = bias[oc];
    for (int ky = 0; ky < 2; ++ky)
      for (int kx = 0; kx < 2; ++kx) {
        const u16* dp = dmap + (((((size_t)b << 12) + (((oh * 2 + ky) << 6) + (ow * 2 + kx)))) << 4);
        const float* wp = wgt + (size_t)oc * 48 + (ky * 2 + kx);
        for (int c = 0; c < 12; ++c)
          acc += bf2f(dp[c]) * wp[(size_t)c * 4];
      }
  }
  out[(((size_t)b << 10) + s) * 64 + oc] = f2bf(acc);
}

// ---------------- Stage 3: scores[b][i][j] (K=64, direct loads) ----------------
__global__ void k_scores(const u16* __restrict__ qt, const u16* __restrict__ kt,
                         float* __restrict__ sc) {
  int b = blockIdx.z;
  int wave = threadIdx.x >> 5, lane = threadIdx.x & 31;
  int m0 = blockIdx.y * 128 + (wave >> 2) * 64;
  int n0 = blockIdx.x * 128 + (wave & 3) * 32;
  const u16* A = qt + (((size_t)b) << 10) * 64;
  const u16* B = kt + (((size_t)b) << 10) * 64;
  v8f acc[4][2] = {};
  for (int kb = 0; kb < 64; kb += 32) {
    v16bf af[4], bfr[2];
    for (int mi = 0; mi < 4; ++mi) af[mi]  = load_a(A + (size_t)(m0 + mi * 16) * 64 + kb, 64);
    for (int ni = 0; ni < 2; ++ni) bfr[ni] = load_b(B + (size_t)(n0 + ni * 16) * 64 + kb, 64);
    for (int mi = 0; mi < 4; ++mi)
      for (int ni = 0; ni < 2; ++ni)
        acc[mi][ni] = wmma_bf16(af[mi], bfr[ni], acc[mi][ni]);
  }
  int nn = lane & 15, mb = (lane >> 4) << 3;
  float* C = sc + (((size_t)b) << 20);
  for (int mi = 0; mi < 4; ++mi)
    for (int ni = 0; ni < 2; ++ni)
      for (int g = 0; g < 8; ++g)
        C[(size_t)(m0 + mi * 16 + mb + g) * 1024 + (n0 + ni * 16 + nn)] = acc[mi][ni][g];
}

// ---------------- Stage 4: row softmax, f32 -> bf16 in place (attn row stride 2048 u16) ----------------
__global__ void k_softmax(float* __restrict__ sc) {
  int row = blockIdx.x;                       // b*1024 + i
  float* p = sc + (size_t)row * 1024;
  u16* o = (u16*)p;
  int t = threadIdx.x;
  float v[4];
  float mx = -3.4e38f;
  for (int j = 0; j < 4; ++j) { v[j] = p[t + j * 256]; mx = fmaxf(mx, v[j]); }
  __shared__ float red[256];
  red[t] = mx; __syncthreads();
  for (int s = 128; s > 0; s >>= 1) { if (t < s) red[t] = fmaxf(red[t], red[t + s]); __syncthreads(); }
  mx = red[0]; __syncthreads();
  float sum = 0.f;
  for (int j = 0; j < 4; ++j) { v[j] = __expf(v[j] - mx); sum += v[j]; }
  red[t] = sum; __syncthreads();
  for (int s = 128; s > 0; s >>= 1) { if (t < s) red[t] += red[t + s]; __syncthreads(); }
  float inv = 1.0f / red[0];
  __syncthreads();
  for (int j = 0; j < 4; ++j) o[t + j * 256] = f2bf(v[j] * inv);
}

// ---------------- Stage 5: v conv — async-LDS double-buffered implicit GEMM (K-tile 64) ----------------
__global__ void k_v_conv(const u16* __restrict__ xb, const u16* __restrict__ vw,
                         const float* __restrict__ v_b, u16* __restrict__ vbuf) {
  __shared__ alignas(16) u16 sA[2][128][72];
  __shared__ alignas(16) u16 sB[2][128][72];
  int b = blockIdx.z;
  int tid = threadIdx.x;
  int wave = tid >> 5, lane = tid & 31;
  int wm = (wave >> 2) * 64, wn = (wave & 3) * 32;
  int m0 = blockIdx.y * 128, n0 = blockIdx.x * 128;
  const u16* A = vw + (size_t)m0 * 1024;
  const u16* xbb = xb + (((size_t)b) << 12) * 256;
  v8f acc[4][2] = {};

  tile_rows_async64(sA[0], A, 1024, 0, tid);
  tile_patch_async64(sB[0], xbb, n0, 0, tid);
  for (int t = 0; t < 16; ++t) {
    int cur = t & 1;
    if (t + 1 < 16) {
      tile_rows_async64(sA[cur ^ 1], A, 1024, (t + 1) * 64, tid);
      tile_patch_async64(sB[cur ^ 1], xbb, n0, (t + 1) * 64, tid);
      wait_async_le8();
    } else {
      wait_async_0();
    }
    __syncthreads();
    for (int ks = 0; ks < 2; ++ks) {
      v16bf af[4], bfr[2];
      for (int mi = 0; mi < 4; ++mi) af[mi]  = lds_frag_a64(sA[cur], wm + mi * 16, ks * 32);
      for (int ni = 0; ni < 2; ++ni) bfr[ni] = lds_frag_b64(sB[cur], wn + ni * 16, ks * 32);
      for (int mi = 0; mi < 4; ++mi)
        for (int ni = 0; ni < 2; ++ni)
          acc[mi][ni] = wmma_bf16(af[mi], bfr[ni], acc[mi][ni]);
    }
    __syncthreads();
  }
  int nn = lane & 15, mb = (lane >> 4) << 3;
  u16* C = vbuf + (((size_t)b) << 20);
  for (int mi = 0; mi < 4; ++mi)
    for (int ni = 0; ni < 2; ++ni)
      for (int g = 0; g < 8; ++g) {
        int m = m0 + wm + mi * 16 + mb + g;
        C[(size_t)m * 1024 + (n0 + wn + ni * 16 + nn)] = f2bf(acc[mi][ni][g] + v_b[m]);
      }
}

// ---------------- Stage 6: out = V * attn^T — async-LDS double-buffered GEMM (K-tile 64) ----------------
__global__ void k_out(const u16* __restrict__ vbuf, const u16* __restrict__ attn,
                      float* __restrict__ out) {
  __shared__ alignas(16) u16 sA[2][128][72];
  __shared__ alignas(16) u16 sB[2][128][72];
  int b = blockIdx.z;
  int tid = threadIdx.x;
  int wave = tid >> 5, lane = tid & 31;
  int wm = (wave >> 2) * 64, wn = (wave & 3) * 32;
  int m0 = blockIdx.y * 128, n0 = blockIdx.x * 128;
  const u16* A = vbuf + (((size_t)b) << 20) + (size_t)m0 * 1024;
  const u16* B = attn + (((size_t)b) << 21) + (size_t)n0 * 2048;  // rows stride 2048 u16
  v8f acc[4][2] = {};

  tile_rows_async64(sA[0], A, 1024, 0, tid);
  tile_rows_async64(sB[0], B, 2048, 0, tid);
  for (int t = 0; t < 16; ++t) {
    int cur = t & 1;
    if (t + 1 < 16) {
      tile_rows_async64(sA[cur ^ 1], A, 1024, (t + 1) * 64, tid);
      tile_rows_async64(sB[cur ^ 1], B, 2048, (t + 1) * 64, tid);
      wait_async_le8();
    } else {
      wait_async_0();
    }
    __syncthreads();
    for (int ks = 0; ks < 2; ++ks) {
      v16bf af[4], bfr[2];
      for (int mi = 0; mi < 4; ++mi) af[mi]  = lds_frag_a64(sA[cur], wm + mi * 16, ks * 32);
      for (int ni = 0; ni < 2; ++ni) bfr[ni] = lds_frag_b64(sB[cur], wn + ni * 16, ks * 32);
      for (int mi = 0; mi < 4; ++mi)
        for (int ni = 0; ni < 2; ++ni)
          acc[mi][ni] = wmma_bf16(af[mi], bfr[ni], acc[mi][ni]);
    }
    __syncthreads();
  }
  int nn = lane & 15, mb = (lane >> 4) << 3;
  float* C = out + (((size_t)b) << 20);
  for (int mi = 0; mi < 4; ++mi)
    for (int ni = 0; ni < 2; ++ni)
      for (int g = 0; g < 8; ++g)
        C[(size_t)(m0 + wm + mi * 16 + mb + g) * 1024 + (n0 + wn + ni * 16 + nn)] = acc[mi][ni][g];
}

extern "C" void kernel_launch(void* const* d_in, const int* in_sizes, int n_in,
                              void* d_out, int out_size, void* d_ws, size_t ws_size,
                              hipStream_t stream) {
  const float* x    = (const float*)d_in[0];
  const float* dc_w = (const float*)d_in[1];
  const float* dc_b = (const float*)d_in[2];
  const float* q_w  = (const float*)d_in[3];
  const float* q_b  = (const float*)d_in[4];
  const float* k_w  = (const float*)d_in[5];
  const float* k_b  = (const float*)d_in[6];
  const float* v_w  = (const float*)d_in[7];
  const float* v_b  = (const float*)d_in[8];
  float* out = (float*)d_out;

  char* ws = (char*)d_ws;
  size_t o = 0;
  u16* xb    = (u16*)(ws + o); o += (size_t)16 * 4096 * 256 * 2;   // BHWC bf16
  u16* dcw   = (u16*)(ws + o); o += (size_t)16 * 2304 * 2;         // dc weights repacked
  u16* vw    = (u16*)(ws + o); o += (size_t)1024 * 1024 * 2;       // v weights repacked
  u16* dmap  = (u16*)(ws + o); o += (size_t)16 * 4096 * 16 * 2;    // dir_map bhwc
  u16* qt    = (u16*)(ws + o); o += (size_t)16 * 1024 * 64 * 2;    // q^T, K-padded 48->64
  u16* kt    = (u16*)(ws + o); o += (size_t)16 * 1024 * 64 * 2;    // k^T, K-padded
  float* sc  = (float*)(ws + o); o += (size_t)16 * 1024 * 1024 * 4; // scores f32 / attn bf16 in place
  u16* vbuf  = (u16*)(ws + o); o += (size_t)16 * 1024 * 1024 * 2;  // v [b][oc][s] bf16

  k_x_to_hwc<<<65536, 256, 0, stream>>>(x, xb);
  k_prep_dcw<<<144, 256, 0, stream>>>(dc_w, dcw);
  k_prep_vw<<<4096, 256, 0, stream>>>(v_w, vw);
  k_dc_conv<<<dim3(32, 1, 16), 256, 0, stream>>>(xb, dcw, dc_b, dmap);
  k_qk_conv<<<8192, 256, 0, stream>>>(dmap, q_w, q_b, k_w, k_b, qt, kt);
  k_scores<<<dim3(8, 8, 16), 256, 0, stream>>>(qt, kt, sc);
  k_softmax<<<16384, 256, 0, stream>>>(sc);
  k_v_conv<<<dim3(8, 8, 16), 256, 0, stream>>>(xb, vw, v_b, vbuf);
  k_out<<<dim3(8, 8, 16), 256, 0, stream>>>(vbuf, (const u16*)sc, out);
}
